// Window3DAttention_8392366096916
// MI455X (gfx1250) — matrix operand order repaired and checked by reading, not compiled, not run on hardware
//
#include <hip/hip_runtime.h>
#include <math.h>

// ---------------------------------------------------------------------------
// Shifted-window 3D attention, fused per-window, CDNA5 (gfx1250, wave32).
//   x: (2,256,16,64,64) f32.  2048 windows of 64 tokens x 256 ch.
//   1 block = 1 window, 8 waves = 8 heads. All intermediates in LDS.
// GEMM path: v_wmma_f32_16x16x32_f16 throughout.
// ---------------------------------------------------------------------------

typedef __attribute__((ext_vector_type(16))) _Float16 v16h;
typedef __attribute__((ext_vector_type(8)))  float    v8f;

struct U32x8 { uint4 lo, hi; };
union FragCast { U32x8 u; v16h h; };
union H8Pack  { _Float16 h[8]; uint4 u; };

__device__ __forceinline__ v16h frag2(const _Float16* plo, const _Float16* phi) {
    FragCast f;
    f.u.lo = *(const uint4*)plo;
    f.u.hi = *(const uint4*)phi;
    return f.h;
}

// A-matrix 16x32 f16 fragment from row-major storage (rows = M, contiguous K).
// lane<16: K[k0..k0+7] + K[k0+16..k0+23]; lane>=16: K[k0+8..] + K[k0+24..].
__device__ __forceinline__ v16h load_a(const _Float16* base, int row0, int stride,
                                       int k0, int l16, int hi) {
    const _Float16* p = base + (row0 + l16) * stride + k0 + (hi ? 8 : 0);
    return frag2(p, p + 16);
}

// B-matrix 32x16 f16 fragment where memory is row-major with rows = N (output
// columns) and contiguous K. lane<16: K[k0..k0+7] + K[k0+16..k0+23]; lane>=16: K[k0+8..] + K[k0+24..].
__device__ __forceinline__ v16h load_b(const _Float16* base, int col0, int stride,
                                       int k0, int l16, int hi) {
    const _Float16* p = base + (col0 + l16) * stride + k0 + (hi ? 8 : 0);
    return frag2(p, p + 16);
}

__device__ __forceinline__ v8f wmma16(v16h a, v16h b, v8f c) {
    return __builtin_amdgcn_wmma_f32_16x16x32_f16(false, a, false, b,
                                                  (short)0, c, false, false);
}

// ------------------------- LDS layout (halfs) ------------------------------
#define XW_STRIDE 264                    // 256 + 8 pad  (also attn_out stride)
#define QK_STRIDE 40                     // 32 + 8 pad
#define VT_STRIDE 72                     // 64 + 8 pad
#define P_STRIDE  72
#define XW_HALFS  (64 * XW_STRIDE)       // 16896
#define HEAD_HALFS 7424                  // q(2560) + k(2560) + vT(2304); P overlays q+k
#define LDS_HALFS (XW_HALFS + 8 * HEAD_HALFS)   // 76288
#define LDS_BYTES (LDS_HALFS * 2)               // 152576 B (< 320 KB WGP LDS)

// ------------------------- prep kernels ------------------------------------
__global__ void gn_stats_kernel(const float* __restrict__ x, float* __restrict__ stats) {
    __shared__ float ssum[256];
    __shared__ float ssq[256];
    const int bg = blockIdx.x;                  // b*32 + g, 64 blocks
    const int tid = threadIdx.x;
    const int b = bg >> 5, g = bg & 31;
    const float* base = x + (((size_t)(b * 256 + g * 8)) << 16);
    float s = 0.f, q = 0.f;
    for (int c = 0; c < 8; ++c) {
        const float* cp = base + (((size_t)c) << 16);
        for (int i = tid; i < 65536; i += 256) {
            float v = cp[i];
            s += v; q += v * v;
        }
    }
    ssum[tid] = s; ssq[tid] = q;
    __syncthreads();
    for (int st = 128; st > 0; st >>= 1) {
        if (tid < st) { ssum[tid] += ssum[tid + st]; ssq[tid] += ssq[tid + st]; }
        __syncthreads();
    }
    if (tid == 0) {
        float mean = ssum[0] * (1.0f / 524288.0f);
        float var  = ssq[0] * (1.0f / 524288.0f) - mean * mean;
        stats[bg]      = mean;
        stats[64 + bg] = rsqrtf(var + 1e-5f);
    }
}

__global__ void cvt_kernel(const float* __restrict__ qkv_w, const float* __restrict__ proj_w,
                           _Float16* __restrict__ wq, _Float16* __restrict__ wp) {
    int i = blockIdx.x * 256 + threadIdx.x;     // 768 blocks -> 196608 threads
    if (i < 196608) wq[i] = (_Float16)qkv_w[i];
    if (i < 65536)  wp[i] = (_Float16)proj_w[i];
}

__global__ void bias_kernel(const float* __restrict__ rpb, float* __restrict__ biasM) {
    int t = blockIdx.x * 256 + threadIdx.x;     // 128 blocks -> 32768 = 8*64*64
    int hh = t >> 12, n = (t >> 6) & 63, m = t & 63;
    int nd = n >> 4, nh = (n >> 2) & 3, nw = n & 3;
    int md = m >> 4, mh = (m >> 2) & 3, mw = m & 3;
    int idx = (nd - md + 3) * 49 + (nh - mh + 3) * 7 + (nw - mw + 3);
    biasM[t] = rpb[idx * 8 + hh];
}

// ------------------------- fused window attention --------------------------
__global__ void __launch_bounds__(256, 1)
win_attn_kernel(const float* __restrict__ x,
                const float* __restrict__ norm_w, const float* __restrict__ norm_b,
                const float* __restrict__ qkv_b,  const float* __restrict__ proj_b,
                const _Float16* __restrict__ wq,  const _Float16* __restrict__ wp,
                const float* __restrict__ biasM,  const float* __restrict__ gnstats,
                float* __restrict__ out) {
    extern __shared__ _Float16 sm[];

    const int tid  = threadIdx.x;
    const int w    = blockIdx.x;                // 0..2047
    const int b    = w >> 10;
    const int widx = w & 1023;
    const int wd = widx >> 8, wh = (widx >> 4) & 15, ww = widx & 15;

    // ---- phase 0: groupnorm + roll(-2) + window partition -> LDS f16 ------
    for (int j = 0; j < 64; ++j) {
        int linear = tid + (j << 8);
        int c = linear >> 6, i = linear & 63;
        int id = i >> 4, ih = (i >> 2) & 3, iw = i & 3;
        int sd = (wd * 4 + id + 2) & 15;
        int sh = (wh * 4 + ih + 2) & 63;
        int sw = (ww * 4 + iw + 2) & 63;
        size_t addr = (((size_t)(b * 256 + c)) << 16) + (sd << 12) + (sh << 6) + sw;
        int g = c >> 3;
        float mu = gnstats[b * 32 + g];
        float is = gnstats[64 + b * 32 + g];
        float v  = (x[addr] - mu) * is * norm_w[c] + norm_b[c];
        sm[i * XW_STRIDE + c] = (_Float16)v;
    }
    __syncthreads();

    const int wave = tid >> 5;                  // head id, 0..7
    const int lane = tid & 31;
    const int l16  = lane & 15;
    const int hi   = lane >> 4;                 // 0 or 1

    _Float16* hb  = sm + XW_HALFS + wave * HEAD_HALFS;
    _Float16* qb  = hb;                         // 64 x QK_STRIDE
    _Float16* kb  = hb + 2560;                  // 64 x QK_STRIDE
    _Float16* vtb = hb + 5120;                  // 32 x VT_STRIDE (v transposed)
    _Float16* pbf = hb;                         // P overlays q+k after S is done

    const float scale = 0.17677669529663687f;   // 1/sqrt(32)

    // ---- phase 1: qkv = xw @ Wqkv^T + b, per-head strips ------------------
    for (int nt = 0; nt < 6; ++nt) {
        const int t3   = nt >> 1;               // 0=q, 1=k, 2=v
        const int ncol = t3 * 256 + wave * 32 + (nt & 1) * 16;
        v8f acc[4] = {};
        for (int kt = 0; kt < 8; ++kt) {
            const int k0 = kt << 5;
            v16h B = load_b(wq, ncol, 256, k0, l16, hi);
#pragma unroll
            for (int mt = 0; mt < 4; ++mt) {
                v16h A = load_a(sm, mt * 16, XW_STRIDE, k0, l16, hi);
                acc[mt] = wmma16(A, B, acc[mt]);
            }
        }
        const float bv   = qkv_b[ncol + l16];
        const int   colw = (nt & 1) * 16 + l16; // within-head column 0..31
#pragma unroll
        for (int mt = 0; mt < 4; ++mt) {
            if (t3 == 2) {                      // v: store transposed, 16B vector
                H8Pack pk;
#pragma unroll
                for (int p = 0; p < 8; ++p) pk.h[p] = (_Float16)(acc[mt][p] + bv);
                *(uint4*)(vtb + colw * VT_STRIDE + mt * 16 + hi * 8) = pk.u;
            } else {                            // q/k: token-major
                _Float16* dst = (t3 == 0) ? qb : kb;
                const float s = (t3 == 0) ? scale : 1.0f;
                const int rowb = mt * 16 + hi * 8;
#pragma unroll
                for (int p = 0; p < 8; ++p)
                    dst[(rowb + p) * QK_STRIDE + colw] = (_Float16)((acc[mt][p] + bv) * s);
            }
        }
    }
    __syncthreads();   // xw dead; its region becomes attn_out below

    // ---- phase 2: S = q k^T + bias, in-register softmax -------------------
    v8f s[4][4];       // [m-tile][n-tile], 128 f32 accumulators
#pragma unroll
    for (int mt = 0; mt < 4; ++mt)
#pragma unroll
        for (int nt = 0; nt < 4; ++nt)
#pragma unroll
            for (int p = 0; p < 8; ++p)
                s[mt][nt][p] = biasM[wave * 4096 + (mt * 16 + hi * 8 + p) * 64 + nt * 16 + l16];

#pragma unroll
    for (int nt = 0; nt < 4; ++nt) {
        v16h Bk = load_b(kb, nt * 16, QK_STRIDE, 0, l16, hi);
#pragma unroll
        for (int mt = 0; mt < 4; ++mt) {
            v16h Aq = load_a(qb, mt * 16, QK_STRIDE, 0, l16, hi);
            s[mt][nt] = wmma16(Aq, Bk, s[mt][nt]);
        }
    }

    // row-wise softmax: row lives at fixed (vgpr p, lane-half), cols across
    // 16 lanes x 4 n-tiles -> xor-shuffles stay inside each 16-lane half.
#pragma unroll
    for (int mt = 0; mt < 4; ++mt) {
#pragma unroll
        for (int p = 0; p < 8; ++p) {
            float m = fmaxf(fmaxf(s[mt][0][p], s[mt][1][p]),
                            fmaxf(s[mt][2][p], s[mt][3][p]));
            m = fmaxf(m, __shfl_xor(m, 1, 32));
            m = fmaxf(m, __shfl_xor(m, 2, 32));
            m = fmaxf(m, __shfl_xor(m, 4, 32));
            m = fmaxf(m, __shfl_xor(m, 8, 32));
            float e0 = __expf(s[mt][0][p] - m);
            float e1 = __expf(s[mt][1][p] - m);
            float e2 = __expf(s[mt][2][p] - m);
            float e3 = __expf(s[mt][3][p] - m);
            float t = e0 + e1 + e2 + e3;
            t += __shfl_xor(t, 1, 32);
            t += __shfl_xor(t, 2, 32);
            t += __shfl_xor(t, 4, 32);
            t += __shfl_xor(t, 8, 32);
            float inv = __frcp_rn(t);
            s[mt][0][p] = e0 * inv; s[mt][1][p] = e1 * inv;
            s[mt][2][p] = e2 * inv; s[mt][3][p] = e3 * inv;
        }
    }

    // store P (f16, token-major); q/k fully consumed -> overlay is safe
#pragma unroll
    for (int mt = 0; mt < 4; ++mt)
#pragma unroll
        for (int nt = 0; nt < 4; ++nt) {
            const int col = nt * 16 + l16;
#pragma unroll
            for (int p = 0; p < 8; ++p)
                pbf[(mt * 16 + hi * 8 + p) * P_STRIDE + col] = (_Float16)s[mt][nt][p];
        }

    // ---- phase 3: O = P @ v  -> attn_out (reuses xw region) ---------------
#pragma unroll
    for (int mt = 0; mt < 4; ++mt) {
#pragma unroll
        for (int dt = 0; dt < 2; ++dt) {
            v8f o = {};
#pragma unroll
            for (int kt = 0; kt < 2; ++kt) {
                v16h Ap = load_a(pbf, mt * 16, P_STRIDE, kt * 32, l16, hi);
                v16h Bv = load_b(vtb, dt * 16, VT_STRIDE, kt * 32, l16, hi);
                o = wmma16(Ap, Bv, o);
            }
            const int col = wave * 32 + dt * 16 + l16;
#pragma unroll
            for (int p = 0; p < 8; ++p)
                sm[(mt * 16 + hi * 8 + p) * XW_STRIDE + col] = (_Float16)o[p];
        }
    }
    __syncthreads();   // all heads written before proj reads full rows

    // ---- phase 4: proj GEMM + bias + residual scatter ---------------------
    for (int nt = 0; nt < 2; ++nt) {
        const int ncol = wave * 32 + nt * 16;
        v8f acc[4] = {};
        for (int kt = 0; kt < 8; ++kt) {
            const int k0 = kt << 5;
            v16h B = load_b(wp, ncol, 256, k0, l16, hi);
#pragma unroll
            for (int mt = 0; mt < 4; ++mt) {
                v16h A = load_a(sm, mt * 16, XW_STRIDE, k0, l16, hi);
                acc[mt] = wmma16(A, B, acc[mt]);
            }
        }
        const int col = ncol + l16;
        const float pbv = proj_b[col];
#pragma unroll
        for (int mt = 0; mt < 4; ++mt) {
#pragma unroll
            for (int p = 0; p < 8; ++p) {
                const int token = mt * 16 + hi * 8 + p;
                const int id = token >> 4, ih = (token >> 2) & 3, iw = token & 3;
                const int sd = (wd * 4 + id + 2) & 15;
                const int sh = (wh * 4 + ih + 2) & 63;
                const int sw = (ww * 4 + iw + 2) & 63;
                size_t addr = (((size_t)(b * 256 + col)) << 16) + (sd << 12) + (sh << 6) + sw;
                out[addr] = x[addr] + pbv + acc[mt][p];
            }
        }
    }
}

// ------------------------- host launcher -----------------------------------
extern "C" void kernel_launch(void* const* d_in, const int* in_sizes, int n_in,
                              void* d_out, int out_size, void* d_ws, size_t ws_size,
                              hipStream_t stream) {
    const float* x      = (const float*)d_in[0];
    const float* norm_w = (const float*)d_in[1];
    const float* norm_b = (const float*)d_in[2];
    const float* qkv_w  = (const float*)d_in[3];
    const float* qkv_b  = (const float*)d_in[4];
    const float* rpb    = (const float*)d_in[5];
    const float* proj_w = (const float*)d_in[6];
    const float* proj_b = (const float*)d_in[7];
    float* out = (float*)d_out;

    // workspace layout (all 16B aligned):
    char* ws = (char*)d_ws;
    float*    stats = (float*)ws;                         //   512 B (64 mean + 64 istd)
    _Float16* wq    = (_Float16*)(ws + 512);              // 393216 B (768x256 f16)
    _Float16* wp    = (_Float16*)(ws + 512 + 393216);     // 131072 B (256x256 f16)
    float*    biasM = (float*)(ws + 512 + 393216 + 131072); // 131072 B (8x64x64 f32)

    gn_stats_kernel<<<64, 256, 0, stream>>>(x, stats);
    cvt_kernel<<<768, 256, 0, stream>>>(qkv_w, proj_w, wq, wp);
    bias_kernel<<<128, 256, 0, stream>>>(rpb, biasM);
    win_attn_kernel<<<2048, 256, LDS_BYTES, stream>>>(
        x, norm_w, norm_b, qkv_b, proj_b, wq, wp, biasM, stats, out);
}
